// TDRegEquiCNN_71734543777977
// MI455X (gfx1250) — matrix operand order repaired and checked by reading, not compiled, not run on hardware
//
#include <hip/hip_runtime.h>
#include <hip/hip_bf16.h>

typedef _Float16 half_t;
typedef __attribute__((ext_vector_type(16))) _Float16 v16h;
typedef __attribute__((ext_vector_type(8)))  _Float16 v8h;
typedef __attribute__((ext_vector_type(8)))  float    v8f;

// ---------------------------------------------------------------------------
// rot90 source-index helper: out[y][x] = in[sy][sx] after r CCW rotations
// ---------------------------------------------------------------------------
__device__ __forceinline__ void rotsrc(int r, int y, int x, int ksz, int& sy, int& sx) {
    switch (r & 3) {
        case 0:  sy = y;            sx = x;            break;
        case 1:  sy = x;            sx = ksz - 1 - y;  break;
        case 2:  sy = ksz - 1 - y;  sx = ksz - 1 - x;  break;
        default: sy = ksz - 1 - x;  sx = y;            break;
    }
}

// ---------------------------------------------------------------------------
// Weight prep: lifted filter  w0:(O,1,k,k) -> [O*4][KKP] f16, zero padded.
// K-layout: per-channel padded to KKP = ceil(k*k/32)*32 (power of two).
// grid: (ceil(KKP/256), O*4); blockIdx.y = output channel n = o*4 + r
// ---------------------------------------------------------------------------
template <int KSZ>
__global__ void prep_lift(const float* __restrict__ w, half_t* __restrict__ out) {
    constexpr int KK  = KSZ * KSZ;
    constexpr int KKP = (KK + 31) & ~31;
    int k = blockIdx.x * blockDim.x + threadIdx.x;
    int n = blockIdx.y;
    if (k >= KKP) return;
    half_t val = (half_t)0.f;
    if (k < KK) {
        int o = n >> 2, r = n & 3;
        int ky = k / KSZ;
        int kx = k - ky * KSZ;
        int sy, sx; rotsrc(r, ky, kx, KSZ, sy, sx);
        val = (half_t)w[o * KK + sy * KSZ + sx];
    }
    out[(size_t)n * KKP + k] = val;
}

// ---------------------------------------------------------------------------
// Weight prep: group filter  w:(O,I,4,k,k) -> [O*4][I*4*KKP] f16, zero padded
// per channel. out[n][ci*KKP + kl] = w[o][i][(g-r)&3][sy][sx] for kl < k*k.
// grid: (ceil(KROW/256), O*4)
// ---------------------------------------------------------------------------
template <int KSZ>
__global__ void prep_group(const float* __restrict__ w, half_t* __restrict__ out,
                           int I, int KROW) {
    constexpr int KK  = KSZ * KSZ;
    constexpr int KKP = (KK + 31) & ~31;      // 32 for 5x5 (power of two)
    int k = blockIdx.x * blockDim.x + threadIdx.x;
    int n = blockIdx.y;
    if (k >= KROW) return;
    int ci = k / KKP;                          // shift
    int kl = k - ci * KKP;                     // mask
    half_t val = (half_t)0.f;
    if (kl < KK) {
        int o = n >> 2, r = n & 3;
        int ky = kl / KSZ;
        int kx = kl - ky * KSZ;
        int i = ci >> 2, g = ci & 3;
        int sy, sx; rotsrc(r, ky, kx, KSZ, sy, sx);
        int gg = (g - r) & 3;
        val = (half_t)w[(((o * I + i) * 4 + gg) * KSZ + sy) * KSZ + sx];
    }
    out[(size_t)n * KROW + k] = val;
}

// ---------------------------------------------------------------------------
// f32 -> f16 elementwise
// ---------------------------------------------------------------------------
__global__ void f32_to_f16(const float* __restrict__ in, half_t* __restrict__ out, int n) {
    int i = blockIdx.x * blockDim.x + threadIdx.x;
    if (i < n) out[i] = (half_t)in[i];
}

// ---------------------------------------------------------------------------
// Implicit-GEMM conv, fully constant-folded, channel-padded K layout.
// GEMM: M = 64*HOUT*WOUT, N = COUT, K = CIN*KKP (KKP chunks align to channels).
// One wave computes a 16(M) x 16*NW(N) tile: one im2col A fragment (gather
// offsets precomputed, loop-invariant; inner loop = 1 load + 1 select per
// element) feeds NW v_wmma_f32_16x16x32_f16 with aligned v16h weight loads.
// NW chosen per layer so COUT tiles exactly (48->3, 96->2x3, 64->4); 24->2 masks 8 of 32 columns.
// ---------------------------------------------------------------------------
template <int CIN, int H, int W, int COUT, int KSZ, int PAD, int NW>
__global__ void __launch_bounds__(256)
conv_wmma(const half_t* __restrict__ in, const half_t* __restrict__ wgt,
          const float* __restrict__ bias, half_t* __restrict__ out) {
    constexpr int HOUT = H + 2 * PAD - KSZ + 1;
    constexpr int WOUT = W + 2 * PAD - KSZ + 1;
    constexpr int HWo  = HOUT * WOUT;
    constexpr int M    = 64 * HWo;
    constexpr int TM   = M / 16;               // exact (batch 64)
    constexpr int KK   = KSZ * KSZ;
    constexpr int KKP  = (KK + 31) & ~31;      // 32 (5x5) or 64 (7x7)
    constexpr int SUBS = KKP / 32;             // chunks per channel
    constexpr int KROW = CIN * KKP;            // padded weight row length
    constexpr int NWID = 16 * NW;              // N columns per wave
    constexpr int TNW  = (COUT + NWID - 1) / NWID;
    constexpr int TILES = TM * TNW;

    const int lane = threadIdx.x & 31;
    const int wave = threadIdx.x >> 5;
    int tile = blockIdx.x * 8 + wave;
    if (tile >= TILES) tile = TILES - 1;       // clamp: duplicate tile, benign

    const int tm = tile / TNW;                 // constant divisor
    const int tn = tile - tm * TNW;

    const int mrow = tm * 16 + (lane & 15);    // A row (16-bit A 16x32 layout)
    const int b  = mrow / HWo;
    const int r0 = mrow - b * HWo;
    const int oy = r0 / WOUT;
    const int ox = r0 - oy * WOUT;
    const int ka = (lane >> 4) * 16;           // A K sub-offset
    const int kb = (lane >> 4) * 16;           // B K sub-offset
    const int nb = tn * NWID + (lane & 15);    // first B column

    // Precompute loop-invariant gather geometry: for each fragment element,
    // its intra-channel k (hence ky,kx,iy,ix) never changes across channels.
    int  offs[SUBS][16];
    bool ok[SUBS][16];
#pragma unroll
    for (int sub = 0; sub < SUBS; ++sub) {
#pragma unroll
        for (int e = 0; e < 16; ++e) {
            int kl = sub * 32 + ka + e;
            int ky = kl / KSZ;
            int kx = kl - ky * KSZ;
            int iy = oy + ky - PAD;
            int ix = ox + kx - PAD;
            bool v = (kl < KK) && ((unsigned)iy < (unsigned)H)
                               && ((unsigned)ix < (unsigned)W);
            ok[sub][e]   = v;
            offs[sub][e] = v ? (b * (CIN * H * W) + iy * W + ix) : 0;
        }
    }

    const half_t* __restrict__ wr[NW];
#pragma unroll
    for (int wv = 0; wv < NW; ++wv) {
        int nn = nb + 16 * wv;
        wr[wv] = wgt + (size_t)(nn < COUT ? nn : 0) * KROW;
    }

    v8f acc[NW];
#pragma unroll
    for (int wv = 0; wv < NW; ++wv) acc[wv] = (v8f){};

#pragma unroll 2
    for (int ci = 0; ci < CIN; ++ci) {
        const int cbase = ci * (H * W);        // wave-uniform channel stride
#pragma unroll
        for (int sub = 0; sub < SUBS; ++sub) {
            v16h afrag;
#pragma unroll
            for (int e = 0; e < 16; ++e) {
                half_t t = in[(size_t)(unsigned)(offs[sub][e] + cbase)];
                afrag[e] = ok[sub][e] ? t : (half_t)0.f;
            }
            const int kc = ci * KKP + sub * 32 + kb;
#pragma unroll
            for (int wv = 0; wv < NW; ++wv) {
                v16h bf = *(const v16h*)(wr[wv] + kc);  // 32B-aligned loads
                acc[wv] = __builtin_amdgcn_wmma_f32_16x16x32_f16(
                    false, afrag, false, bf, (short)0, acc[wv], false, false);
            }
        }
        if (ci + 1 < CIN)   // speculative prefetch of next channel (global_prefetch_b8)
            __builtin_prefetch(in + (size_t)(unsigned)(offs[0][0] + cbase) + (H * W), 0, 0);
    }

    const int mhi = (lane >> 4) << 3;
#pragma unroll
    for (int v = 0; v < 8; ++v) {
        int m  = tm * 16 + v + mhi;            // C layout: M = v (+8 hi lanes)
        int bb = m / HWo;
        int rr = m - bb * HWo;
        int yy = rr / WOUT;
        int xx = rr - yy * WOUT;
        size_t base = (size_t)bb * COUT * HWo + yy * WOUT + xx;
#pragma unroll
        for (int wv = 0; wv < NW; ++wv) {
            int nn = nb + 16 * wv;
            if (nn < COUT) {
                float val = acc[wv][v] + bias[nn >> 2];  // bias repeated G=4
                out[base + (size_t)nn * HWo] = (half_t)(val > 0.f ? val : 0.f);
            }
        }
    }
}

// ---------------------------------------------------------------------------
// 2x2 max pool, stride 2, f16, constant dims
// ---------------------------------------------------------------------------
template <int C, int H, int W>
__global__ void pool_kernel(const half_t* __restrict__ in, half_t* __restrict__ out) {
    constexpr int Ho = H / 2, Wo = W / 2;
    constexpr int TOTAL = 64 * C * Ho * Wo;
    int idx = blockIdx.x * blockDim.x + threadIdx.x;
    if (idx >= TOTAL) return;
    int xo = idx % Wo; int t = idx / Wo;
    int yo = t % Ho;   t /= Ho;                // t = b*C + c
    const half_t* p = in + ((size_t)t * H + yo * 2) * W + xo * 2;
    float m = fmaxf(fmaxf((float)p[0], (float)p[1]),
                    fmaxf((float)p[W], (float)p[W + 1]));
    out[idx] = (half_t)m;
}

// ---------------------------------------------------------------------------
// Global average pool over HW per (b,c); f16 in, f16 out row-major [64][C]
// ---------------------------------------------------------------------------
template <int C, int HW>
__global__ void gap_kernel(const half_t* __restrict__ in, half_t* __restrict__ out) {
    int idx = blockIdx.x * blockDim.x + threadIdx.x;
    if (idx >= 64 * C) return;
    const half_t* p = in + (size_t)idx * HW;
    float s = 0.f;
    for (int i = 0; i < HW; ++i) s += (float)p[i];
    out[idx] = (half_t)(s * (1.f / (float)HW));
}

// ---------------------------------------------------------------------------
// FC GEMM via WMMA: Y[m][n] = sum_k A[m][k]*W[n][k] + bias[n]; M = 64.
// One wave per 16x16 tile; A/B fragments are aligned vector loads (K%32==0).
// ---------------------------------------------------------------------------
template <int NT, int KDIM, bool RELU, bool OUTF32>
__global__ void __launch_bounds__(32)
gemm_wmma_fc(const half_t* __restrict__ A, const half_t* __restrict__ Wt,
             const float* __restrict__ bias, void* __restrict__ outp, int N) {
    const int lane = threadIdx.x;
    const int tile = blockIdx.x;
    const int tm = tile / NT;
    const int tn = tile - tm * NT;
    const int mrow = tm * 16 + (lane & 15);
    const int n    = tn * 16 + (lane & 15);
    const int ka = (lane >> 4) * 16;
    const int kb = (lane >> 4) * 16;
    const half_t* __restrict__ Arow = A  + (size_t)mrow * KDIM;
    const half_t* __restrict__ Wrow = Wt + (size_t)(n < N ? n : 0) * KDIM;

    v8f acc = {};
#pragma unroll
    for (int kc = 0; kc < KDIM; kc += 32) {
        v8h alo = *(const v8h*)(Arow + kc + ka);
        v8h ahi = *(const v8h*)(Arow + kc + ka + 8);
        v16h afrag;
#pragma unroll
        for (int e = 0; e < 8; ++e) { afrag[e] = alo[e]; afrag[e + 8] = ahi[e]; }
        v16h bfrag = *(const v16h*)(Wrow + kc + kb);
        acc = __builtin_amdgcn_wmma_f32_16x16x32_f16(
            false, afrag, false, bfrag, (short)0, acc, false, false);
    }
    if (n < N) {
        float bv = bias[n];
#pragma unroll
        for (int v = 0; v < 8; ++v) {
            int m = tm * 16 + v + ((lane >> 4) << 3);
            float val = acc[v] + bv;
            if (RELU) val = val > 0.f ? val : 0.f;
            if (OUTF32) ((float*)outp)[(size_t)m * N + n] = val;
            else        ((half_t*)outp)[(size_t)m * N + n] = (half_t)val;
        }
    }
}

// ---------------------------------------------------------------------------
// Host launcher
// ---------------------------------------------------------------------------
extern "C" void kernel_launch(void* const* d_in, const int* in_sizes, int n_in,
                              void* d_out, int out_size, void* d_ws, size_t ws_size,
                              hipStream_t stream) {
    const float* x   = (const float*)d_in[0];
    const float* w0  = (const float*)d_in[1];
    const float* b0  = (const float*)d_in[2];
    const float* w1  = (const float*)d_in[3];
    const float* b1  = (const float*)d_in[4];
    const float* w2  = (const float*)d_in[5];
    const float* b2  = (const float*)d_in[6];
    const float* w3  = (const float*)d_in[7];
    const float* b3  = (const float*)d_in[8];
    const float* w4  = (const float*)d_in[9];
    const float* b4  = (const float*)d_in[10];
    const float* w5  = (const float*)d_in[11];
    const float* b5  = (const float*)d_in[12];
    const float* fw1 = (const float*)d_in[13];
    const float* fb1 = (const float*)d_in[14];
    const float* fw2 = (const float*)d_in[15];
    const float* fb2 = (const float*)d_in[16];

    // workspace bump allocator (256B aligned)
    char* ws = (char*)d_ws;
    size_t off = 0;
    auto take = [&](size_t elems) {
        half_t* p = (half_t*)(ws + off);
        off += (elems * sizeof(half_t) + 255) & ~(size_t)255;
        return p;
    };
    half_t* bufB  = take(26001408);    // max: act1 = 64*48*92*92
    half_t* bufA  = take(13000704);    // max: act0/act3
    half_t* x16   = take(589824);      // 64*1*96*96
    half_t* wL0   = take(24 * 64);     // KROW = 1*64
    half_t* wG1   = take(48 * 768);    // KROW = 24*32
    half_t* wG2   = take(48 * 1536);   // KROW = 48*32
    half_t* wG3   = take(96 * 1536);
    half_t* wG4   = take(96 * 3072);   // KROW = 96*32
    half_t* wG5   = take(64 * 3072);
    half_t* fw1h  = take(64 * 64);
    half_t* fw2h  = take(10 * 64);
    half_t* gap16 = take(64 * 64);
    half_t* fc1o  = take(64 * 64);

    // --- input conversion + weight prep (channel-padded K layout) ---
    f32_to_f16<<<(589824 + 255) / 256, 256, 0, stream>>>(x, x16, 589824);
    prep_lift<7><<<dim3(1, 24), 256, 0, stream>>>(w0, wL0);
    prep_group<5><<<dim3(3, 48), 256, 0, stream>>>(w1, wG1, 6, 768);
    prep_group<5><<<dim3(6, 48), 256, 0, stream>>>(w2, wG2, 12, 1536);
    prep_group<5><<<dim3(6, 96), 256, 0, stream>>>(w3, wG3, 12, 1536);
    prep_group<5><<<dim3(12, 96), 256, 0, stream>>>(w4, wG4, 24, 3072);
    prep_group<5><<<dim3(12, 64), 256, 0, stream>>>(w5, wG5, 24, 3072);
    f32_to_f16<<<(4096 + 255) / 256, 256, 0, stream>>>(fw1, fw1h, 4096);
    f32_to_f16<<<(640 + 255) / 256, 256, 0, stream>>>(fw2, fw2h, 640);

    auto nblocks = [](int Hout, int Wout, int Cout, int NW) {
        int nwid = 16 * NW;
        int tiles = (64 * Hout * Wout / 16) * ((Cout + nwid - 1) / nwid);
        return (tiles + 7) / 8;
    };

    // --- pipeline (ping-pong bufA <-> bufB) ---
    conv_wmma<1, 96, 96, 24, 7, 1, 2>
        <<<nblocks(92, 92, 24, 2), 256, 0, stream>>>(x16, wL0, b0, bufA);  // (64,24,92,92)
    conv_wmma<24, 92, 92, 48, 5, 2, 3>
        <<<nblocks(92, 92, 48, 3), 256, 0, stream>>>(bufA, wG1, b1, bufB); // (64,48,92,92)
    pool_kernel<48, 92, 92>
        <<<(64 * 48 * 46 * 46 + 255) / 256, 256, 0, stream>>>(bufB, bufA); // (64,48,46,46)
    conv_wmma<48, 46, 46, 48, 5, 2, 3>
        <<<nblocks(46, 46, 48, 3), 256, 0, stream>>>(bufA, wG2, b2, bufB); // (64,48,46,46)
    conv_wmma<48, 46, 46, 96, 5, 2, 3>
        <<<nblocks(46, 46, 96, 3), 256, 0, stream>>>(bufB, wG3, b3, bufA); // (64,96,46,46)
    pool_kernel<96, 46, 46>
        <<<(64 * 96 * 23 * 23 + 255) / 256, 256, 0, stream>>>(bufA, bufB); // (64,96,23,23)
    conv_wmma<96, 23, 23, 96, 5, 2, 3>
        <<<nblocks(23, 23, 96, 3), 256, 0, stream>>>(bufB, wG4, b4, bufA); // (64,96,23,23)
    conv_wmma<96, 23, 23, 64, 5, 1, 4>
        <<<nblocks(21, 21, 64, 4), 256, 0, stream>>>(bufA, wG5, b5, bufB); // (64,64,21,21)

    gap_kernel<64, 441><<<16, 256, 0, stream>>>(bufB, gap16);

    // FC1: (64,64)@(64,64)^T + fb1, relu -> f16 ; FC2: -> f32 d_out (64,10)
    gemm_wmma_fc<4, 64, true, false><<<16, 32, 0, stream>>>(gap16, fw1h, fb1, fc1o, 64);
    gemm_wmma_fc<1, 64, false, true><<<4, 32, 0, stream>>>(fc1o, fw2h, fb2, d_out, 10);
}
